// GraphConvolution_23278722744980
// MI455X (gfx1250) — matrix operand order repaired and checked by reading, not compiled, not run on hardware
//
#include <hip/hip_runtime.h>

typedef __attribute__((ext_vector_type(16))) __bf16       v16bf;
typedef __attribute__((ext_vector_type(8)))  float        v8f;
typedef __attribute__((ext_vector_type(8)))  unsigned int v8u;

#define N_NODES 50000
#define N_EDGES 400000
#define D_IN    1024
#define D_OUT   1024

#define BM 128
#define BN 128
#define BK 32
#define NCHUNK (D_IN / BK)   // 32 K-chunks
// Fragment-major LDS tile: [group g = kk>>2][row (m or n), 128][j = kk&3]
// group stride padded 512 -> 516 u32 (keeps 16B alignment, breaks 64-bank wrap)
#define GSTR   516
#define REGION (4 * GSTR)    // one matrix buffer (Ahi/Alo/Bhi/Blo) = 2064 u32
#define BUFU   (4 * REGION)  // Ahi | Alo | Bhi | Blo = 8256 u32 (33 KB)

__device__ __forceinline__ unsigned int pk2(__bf16 a, __bf16 b) {
    unsigned short ua = __builtin_bit_cast(unsigned short, a);
    unsigned short ub = __builtin_bit_cast(unsigned short, b);
    return (unsigned int)ua | ((unsigned int)ub << 16);
}

// LDS byte offset of a generic pointer into __shared__ (flat LDS aperture:
// addr[31:0] is the wave-relative LDS byte address).
__device__ __forceinline__ unsigned int ldso(const unsigned int* p) {
    return (unsigned int)(size_t)(const void*)p;
}

// CDNA5 async copy global -> LDS (tracked by ASYNCcnt), 16B per lane.
__device__ __forceinline__ void async_load_b128(unsigned int lds_byte,
                                                const unsigned int* gaddr) {
    asm volatile("global_load_async_to_lds_b128 %0, %1, off"
                 :: "v"(lds_byte), "v"(gaddr)
                 : "memory");
}
__device__ __forceinline__ void wait_async0() {
    asm volatile("s_wait_asynccnt 0" ::: "memory");
}

// Two aligned b128 LDS loads -> one 8-VGPR WMMA operand.
__device__ __forceinline__ v8u frag8(const unsigned int* region, int g0, int g1,
                                     int idx4) {
    uint4 q0 = *(const uint4*)(region + g0 * GSTR + idx4);
    uint4 q1 = *(const uint4*)(region + g1 * GSTR + idx4);
    v8u f;
    f[0] = q0.x; f[1] = q0.y; f[2] = q0.z; f[3] = q0.w;
    f[4] = q1.x; f[5] = q1.y; f[6] = q1.z; f[7] = q1.w;
    return f;
}

// ---------------------------------------------------------------------------
// Pre-pack W (fp32 [1024][1024]) into bf16 hi/lo, fragment-major global image:
// pack column kk = pack(bf16(W[2kk][n]), bf16(W[2kk+1][n])) stored at
//   [(chunk*4 + (kk&15)>>2) * 1024 + n] * 4 + (kk&3)
// so each K-chunk's B tile is four contiguous 2KB-per-128-cols group slices.
// ---------------------------------------------------------------------------
__global__ void prep_w_kernel(const float* __restrict__ W,
                              unsigned int* __restrict__ wfh,
                              unsigned int* __restrict__ wfl) {
    int i   = blockIdx.x * blockDim.x + threadIdx.x;  // 0 .. 512*1024-1
    int kkg = i >> 10;                                // global packed-K row
    int n   = i & 1023;
    float w0 = W[(size_t)(2 * kkg)     * D_OUT + n];
    float w1 = W[(size_t)(2 * kkg + 1) * D_OUT + n];
    __bf16 h0 = (__bf16)w0, h1 = (__bf16)w1;
    __bf16 l0 = (__bf16)(w0 - (float)h0);
    __bf16 l1 = (__bf16)(w1 - (float)h1);
    int c = kkg >> 4, kkl = kkg & 15;
    int g = kkl >> 2, j = kkl & 3;
    size_t o = ((size_t)(c * 4 + g) * 1024 + n) * 4 + j;
    wfh[o] = pk2(h0, h1);
    wfl[o] = pk2(l0, l1);
}

// ---------------------------------------------------------------------------
// Staging helpers
// ---------------------------------------------------------------------------
__device__ __forceinline__ void stage_B_async(unsigned int* buf,   // LDS buffer base
                                              const unsigned int* __restrict__ wfh,
                                              const unsigned int* __restrict__ wfl,
                                              int c, int n0, int t) {
    // copy 4 group-slices of 512 u32 (hi and lo), 4 async b128 per thread
    int g   = t >> 6;             // 64 threads per group
    int rem = (t & 63) * 8;       // 0..504
    size_t src = (size_t)(c * 4 + g) * 4096 + (size_t)n0 * 4 + rem;
    unsigned int* dh = buf + 2 * REGION + g * GSTR + rem;   // B-hi region
    unsigned int* dl = buf + 3 * REGION + g * GSTR + rem;   // B-lo region
    async_load_b128(ldso(dh),     wfh + src);
    async_load_b128(ldso(dh + 4), wfh + src + 4);
    async_load_b128(ldso(dl),     wfl + src);
    async_load_b128(ldso(dl + 4), wfl + src + 4);
}

__device__ __forceinline__ void load_A(float4 xa[4], const float* __restrict__ x,
                                       int m0, int kb, int t) {
    #pragma unroll
    for (int p = 0; p < 4; ++p) {
        int m  = (t >> 3) + p * 32;
        int k4 = (t & 7) * 4;
        int gm = m0 + m;
        xa[p] = make_float4(0.f, 0.f, 0.f, 0.f);
        if (gm < N_NODES)
            xa[p] = *(const float4*)(x + (size_t)gm * D_IN + kb + k4);
    }
}

__device__ __forceinline__ void store_A(unsigned int* buf, const float4 xa[4], int t) {
    const int q = t & 7;          // K-quad owner: pack cols kk2=2q, 2q+1
    const int g = q >> 1;
    const int j = (q & 1) * 2;    // {0,2}: b64 covers j, j+1
    #pragma unroll
    for (int p = 0; p < 4; ++p) {
        int m = (t >> 3) + p * 32;
        float4 xv = xa[p];
        __bf16 h0 = (__bf16)xv.x, h1 = (__bf16)xv.y;
        __bf16 h2 = (__bf16)xv.z, h3 = (__bf16)xv.w;
        __bf16 e0 = (__bf16)(xv.x - (float)h0), e1 = (__bf16)(xv.y - (float)h1);
        __bf16 e2 = (__bf16)(xv.z - (float)h2), e3 = (__bf16)(xv.w - (float)h3);
        uint2 hv = make_uint2(pk2(h0, h1), pk2(h2, h3));
        uint2 lv = make_uint2(pk2(e0, e1), pk2(e2, e3));
        *(uint2*)(buf + g * GSTR + m * 4 + j)          = hv;   // A-hi
        *(uint2*)(buf + REGION + g * GSTR + m * 4 + j) = lv;   // A-lo
    }
}

// ---------------------------------------------------------------------------
// GEMM: h = x @ W via bf16 hi/lo split, V_WMMA_F32_16X16X32_BF16.
// 256 threads = 8 waves; block tile 128x128; wave tile 32x64.
// Double-buffered LDS; B staged with global_load_async_to_lds_b128.
// ---------------------------------------------------------------------------
__global__ __launch_bounds__(256)
void gemm_kernel(const float* __restrict__ x,
                 const unsigned int* __restrict__ wfh,
                 const unsigned int* __restrict__ wfl,
                 float* __restrict__ h) {
    __shared__ unsigned int lds[2][BUFU];

    const int t    = threadIdx.x;
    const int n0   = blockIdx.x * BN;
    const int m0   = blockIdx.y * BM;
    const int lane = t & 31;
    const int half = lane >> 4;
    const int l16  = lane & 15;
    const int wid  = t >> 5;
    const int m_off = (wid & 3) * 32;   // wave M origin in block tile
    const int n_off = (wid >> 2) * 64;  // wave N origin in block tile

    v8f acc[2][4];
    #pragma unroll
    for (int ms = 0; ms < 2; ++ms)
        #pragma unroll
        for (int ns = 0; ns < 4; ++ns)
            acc[ms][ns] = {};

    float4 xa[4];

    // ---- prologue: stage chunk 0 into lds[0] ----
    stage_B_async(lds[0], wfh, wfl, 0, n0, t);
    load_A(xa, x, m0, 0, t);
    store_A(lds[0], xa, t);
    wait_async0();
    __syncthreads();

    for (int c = 0; c < NCHUNK; ++c) {
        unsigned int* L    = lds[c & 1];
        unsigned int* Lnxt = lds[(c + 1) & 1];
        const bool has_next = (c + 1) < NCHUNK;

        // ---- prefetch chunk c+1 (overlaps with WMMA below) ----
        if (has_next) {
            stage_B_async(Lnxt, wfh, wfl, c + 1, n0, t);
            load_A(xa, x, m0, (c + 1) * BK, t);
        }

        // ---- fragment loads: 2x ds_load_b128 per operand ----
        // A (ISA 16-bit A 16x32): v0..3 -> kk = j + 4*half  (group half)
        //                         v4..7 -> kk = 8+j + 4*half (group 2+half)
        // B (same as A):          v0..3 -> kk = j + 4*half, v4..7 -> kk = 8+j + 4*half (groups h, 2+h)
        const unsigned int* LAh = L;
        const unsigned int* LAl = L + REGION;
        const unsigned int* LBh = L + 2 * REGION;
        const unsigned int* LBl = L + 3 * REGION;
        v8u ah[2], al[2], bh[4], bl[4];
        #pragma unroll
        for (int ms = 0; ms < 2; ++ms) {
            int ai = (m_off + ms * 16 + l16) * 4;
            ah[ms] = frag8(LAh, half, 2 + half, ai);
            al[ms] = frag8(LAl, half, 2 + half, ai);
        }
        #pragma unroll
        for (int ns = 0; ns < 4; ++ns) {
            int bi = (n_off + ns * 16 + l16) * 4;
            bh[ns] = frag8(LBh, half, 2 + half, bi);
            bl[ns] = frag8(LBl, half, 2 + half, bi);
        }

        // ---- 3-product bf16 split accumulation (24 WMMAs) ----
        #pragma unroll
        for (int ms = 0; ms < 2; ++ms) {
            v16bf Ah = __builtin_bit_cast(v16bf, ah[ms]);
            v16bf Al = __builtin_bit_cast(v16bf, al[ms]);
            #pragma unroll
            for (int ns = 0; ns < 4; ++ns) {
                v16bf Bh = __builtin_bit_cast(v16bf, bh[ns]);
                v16bf Bl = __builtin_bit_cast(v16bf, bl[ns]);
                v8f cc = acc[ms][ns];
                cc = __builtin_amdgcn_wmma_f32_16x16x32_bf16(false, Ah, false, Bh, (short)0, cc, false, false);
                cc = __builtin_amdgcn_wmma_f32_16x16x32_bf16(false, Ah, false, Bl, (short)0, cc, false, false);
                cc = __builtin_amdgcn_wmma_f32_16x16x32_bf16(false, Al, false, Bh, (short)0, cc, false, false);
                acc[ms][ns] = cc;
            }
        }

        // ---- finish staging chunk c+1, then one barrier per iteration ----
        if (has_next) {
            store_A(Lnxt, xa, t);   // compiler waits LOADcnt for xa as needed
            wait_async0();          // our async B writes are in LDS
        }
        __syncthreads();
    }

    // ---- epilogue: C/D layout VGPR v -> row v + 8*half, col = lane&15 ----
    #pragma unroll
    for (int ms = 0; ms < 2; ++ms)
        #pragma unroll
        for (int ns = 0; ns < 4; ++ns)
            #pragma unroll
            for (int v = 0; v < 8; ++v) {
                int row = m0 + m_off + ms * 16 + v + half * 8;
                int col = n0 + n_off + ns * 16 + l16;
                if (row < N_NODES)
                    h[(size_t)row * D_OUT + col] = acc[ms][ns][v];
            }
}

// ---------------------------------------------------------------------------
// out[r][d] = bias[d]  (broadcast init; d_out is poisoned by the harness)
// ---------------------------------------------------------------------------
__global__ void init_out_kernel(const float4* __restrict__ bias4,
                                float4* __restrict__ out4) {
    size_t i = (size_t)blockIdx.x * blockDim.x + threadIdx.x;
    out4[i] = bias4[i & 255];
}

// ---------------------------------------------------------------------------
// SpMM scatter: one block per edge, hardware f32 global atomics (no-return).
// ---------------------------------------------------------------------------
__global__ __launch_bounds__(256)
void spmm_kernel(const float* __restrict__ h,
                 const float* __restrict__ vals,
                 const int* __restrict__ rows,
                 const int* __restrict__ cols,
                 float* __restrict__ out) {
    const int e = blockIdx.x;
    const int r = rows[e];
    const int c = cols[e];
    const float v = vals[e];
    const int d = threadIdx.x * 4;
    float4 hv = *(const float4*)(h + (size_t)c * D_OUT + d);
    float* o = out + (size_t)r * D_OUT + d;
    __hip_atomic_fetch_add(o + 0, v * hv.x, __ATOMIC_RELAXED, __HIP_MEMORY_SCOPE_AGENT);
    __hip_atomic_fetch_add(o + 1, v * hv.y, __ATOMIC_RELAXED, __HIP_MEMORY_SCOPE_AGENT);
    __hip_atomic_fetch_add(o + 2, v * hv.z, __ATOMIC_RELAXED, __HIP_MEMORY_SCOPE_AGENT);
    __hip_atomic_fetch_add(o + 3, v * hv.w, __ATOMIC_RELAXED, __HIP_MEMORY_SCOPE_AGENT);
}

extern "C" void kernel_launch(void* const* d_in, const int* in_sizes, int n_in,
                              void* d_out, int out_size, void* d_ws, size_t ws_size,
                              hipStream_t stream) {
    const float* x    = (const float*)d_in[0];
    const float* W    = (const float*)d_in[1];
    const float* bias = (const float*)d_in[2];
    const float* vals = (const float*)d_in[3];
    const int*   rows = (const int*)d_in[4];
    const int*   cols = (const int*)d_in[5];
    float* out = (float*)d_out;

    // workspace layout: h (204.8 MB) | Wfrag_hi (2 MB) | Wfrag_lo (2 MB)
    char* ws = (char*)d_ws;
    float* h = (float*)ws;
    unsigned int* wfh = (unsigned int*)(ws + (size_t)N_NODES * D_OUT * sizeof(float));
    unsigned int* wfl = wfh + (size_t)(D_IN / 2) * D_OUT;

    prep_w_kernel<<<(D_IN / 2) * D_OUT / 256, 256, 0, stream>>>(W, wfh, wfl);

    dim3 grid(D_OUT / BN, (N_NODES + BM - 1) / BM);
    gemm_kernel<<<grid, 256, 0, stream>>>(x, wfh, wfl, h);

    init_out_kernel<<<(size_t)N_NODES * (D_OUT / 4) / 256, 256, 0, stream>>>(
        (const float4*)bias, (float4*)out);

    spmm_kernel<<<N_EDGES, D_OUT / 4, 0, stream>>>(h, vals, rows, cols, out);
}
